// AttentionCpp_62182536511490
// MI455X (gfx1250) — matrix operand order repaired and checked by reading, not compiled, not run on hardware
//
#include <hip/hip_runtime.h>

typedef __bf16 bf16_t;
typedef __attribute__((ext_vector_type(16))) __bf16 v16bf;
typedef __attribute__((ext_vector_type(8)))  __bf16 v8bf;
typedef __attribute__((ext_vector_type(8)))  float  v8f;
typedef unsigned int u32x4 __attribute__((ext_vector_type(4)));
typedef int          i32x4 __attribute__((ext_vector_type(4)));
typedef int          i32x8 __attribute__((ext_vector_type(8)));

#define SEQ 2048
#define CX  768
#define NH  16
#define DH  48
#define DP  64   // padded head dim
#define NEG_BIG (-1e9f)

// ---------------- WMMA helpers (CDNA5 gfx1250, wave32) ----------------

__device__ inline v8f wmma_bf16(v16bf a, v16bf b, v8f c) {
  // D = A(16x32 bf16) * B(32x16 bf16) + C(16x16 f32)
  return __builtin_amdgcn_wmma_f32_16x16x32_bf16(
      /*neg_a=*/false, a, /*neg_b=*/false, b,
      /*c_mod=*/(short)0, c, /*reuse_a=*/false, /*reuse_b=*/false);
}

// A fragment: base points at tile [rows][ld] (row-major, bf16).
// Lane l holds row (l&15). K layout: element j <-> K = half*16+j.
__device__ inline v16bf load_a_frag(const bf16_t* base, int ld, int kbase, int lane) {
  int r = lane & 15, gh = lane >> 4;
  const bf16_t* p = base + r * ld + kbase + gh * 16;
  v8bf lo = *(const v8bf*)(p);
  v8bf hi = *(const v8bf*)(p + 8);
  return __builtin_shufflevector(lo, hi, 0,1,2,3,4,5,6,7,8,9,10,11,12,13,14,15);
}

// B fragment: base points at K-transposed tile [n][ldk] so each lane reads
// 16 contiguous K values: lane holds col n=(l&15), element j <-> K = half*16+j.
__device__ inline v16bf load_b_frag(const bf16_t* base, int ld, int kbase, int lane) {
  int n = lane & 15, gh = lane >> 4;
  const bf16_t* p = base + n * ld + kbase + gh * 16;
  v8bf lo = *(const v8bf*)(p);
  v8bf hi = *(const v8bf*)(p + 8);
  return __builtin_shufflevector(lo, hi, 0,1,2,3,4,5,6,7,8,9,10,11,12,13,14,15);
}

// ---------------- TDM: 2D tile DMA global -> LDS (bf16 elements) ----------------
// Builds a D# per CDNA5 ISA and issues TENSOR_LOAD_TO_LDS.
// tile_cols = tile_dim0 (contiguous), tile_rows = tile_dim1,
// row_stride_elems = tensor_dim0_stride. Units: elements (data_size = 2 bytes).

__device__ inline unsigned lds_off_u32(const void* p) {
  return (unsigned)(unsigned long long)p;   // LDS aperture: addr[31:0] is the LDS offset
}

__device__ inline void tdm_load_tile_2d(unsigned lds_byte_off, const void* gptr,
                                        unsigned tile_cols, unsigned tile_rows,
                                        unsigned row_stride_elems) {
  unsigned long long ga = (unsigned long long)gptr;
  u32x4 g0;
  g0[0] = 1u;                                             // count=1 (valid), load, no gather
  g0[1] = lds_byte_off;                                   // lds_addr
  g0[2] = (unsigned)ga;                                   // global_addr[31:0]
  g0[3] = (unsigned)((ga >> 32) & 0x01FFFFFFu)            // global_addr[56:32]
          | 0x80000000u;                                  // type=2 ("image")
  i32x8 g1;
  g1[0] = 1 << 16;                                        // data_size=1 (2 bytes), wg_mask=0
  g1[1] = (int)(tile_cols << 16);                         // tensor_dim0[15:0]
  g1[2] = (int)(tile_rows << 16);                         // tensor_dim0[31:16]=0 | tensor_dim1[15:0]
  g1[3] = (int)(tile_cols << 16);                         // tensor_dim1[31:16]=0 | tile_dim0
  g1[4] = (int)tile_rows;                                 // tile_dim1 | tile_dim2=0
  g1[5] = (int)row_stride_elems;                          // tensor_dim0_stride[31:0]
  g1[6] = 0;                                              // stride0 hi | stride1 lo
  g1[7] = 0;
  i32x4 z4 = {0, 0, 0, 0};
#if __clang_major__ >= 23
  i32x8 z8 = {0, 0, 0, 0, 0, 0, 0, 0};
  __builtin_amdgcn_tensor_load_to_lds(g0, g1, z4, z4, z8, 0);
#else
  __builtin_amdgcn_tensor_load_to_lds(g0, g1, z4, z4, 0);
#endif
}

// ---------------- Kernel 0: fp32 -> bf16 conversion + zero pads ----------------

__global__ __launch_bounds__(256) void prep_kernel(
    const float* __restrict__ x,
    const float* __restrict__ Wq, const float* __restrict__ Wk,
    const float* __restrict__ Wv, const float* __restrict__ Wg,
    bf16_t* __restrict__ xb, bf16_t* __restrict__ wb,
    bf16_t* __restrict__ qb, bf16_t* __restrict__ kb, bf16_t* __restrict__ vb) {
  int i = blockIdx.x * 256 + threadIdx.x;          // launched with exactly 4*WSZ threads
  const int WSZ = CX * CX;                         // 589824
  if (i < 4 * WSZ) {
    int w = i / WSZ, off = i - w * WSZ;
    float v = (w == 0) ? Wq[off] : (w == 1) ? Wk[off] : (w == 2) ? Wv[off] : Wg[off];
    wb[i] = (bf16_t)v;
  }
  if (i < SEQ * CX) xb[i] = (bf16_t)x[i];
  if (i < SEQ * NH * DP) {                         // zero padded q/k/v (cols 48..63 stay 0)
    qb[i] = (bf16_t)0.f; kb[i] = (bf16_t)0.f; vb[i] = (bf16_t)0.f;
  }
}

// ---------------- Kernel 1: fused QKVG projections ----------------
// grid = (SEQ/128, CX/64, 4). Block tile 128x64, 8 waves of 32x32 each.
// A tile (x) staged by TDM; B tile (W) staged manually transposed.

__global__ __launch_bounds__(256) void proj_kernel(
    const bf16_t* __restrict__ xb, const bf16_t* __restrict__ wb,
    const float* __restrict__ bq,
    bf16_t* __restrict__ qb, bf16_t* __restrict__ kb, bf16_t* __restrict__ vb,
    float* __restrict__ gate) {
  const int rt = blockIdx.x * 128, ct = blockIdx.y * 64, w = blockIdx.z;
  const bf16_t* W = wb + (size_t)w * CX * CX;

  __shared__ __align__(16) bf16_t xs[128][32];   // A tile, row-major (TDM target)
  __shared__ __align__(16) bf16_t wsT[64][32];   // B tile, transposed [n][k]

  const int tid = threadIdx.x, lane = tid & 31, wid = tid >> 5;
  const int wr = (wid >> 1) * 32, wc = (wid & 1) * 32;

  v8f acc[2][2];
#pragma unroll
  for (int a = 0; a < 2; ++a)
#pragma unroll
    for (int b = 0; b < 2; ++b)
#pragma unroll
      for (int i = 0; i < 8; ++i) acc[a][b][i] = 0.f;

  for (int k0 = 0; k0 < CX; k0 += 32) {
    if (wid == 0) {  // DMA x tile: 128 rows x 32 cols, row pitch CX
      tdm_load_tile_2d(lds_off_u32(&xs[0][0]),
                       xb + (size_t)rt * CX + k0, 32u, 128u, (unsigned)CX);
    }
    { // stage W tile transposed: read [k][n] coalesced, write [n][k]
      int kk = tid >> 3, n0 = (tid & 7) * 8;
      v8bf wv = *(const v8bf*)(W + (size_t)(k0 + kk) * CX + ct + n0);
#pragma unroll
      for (int j = 0; j < 8; ++j) wsT[n0 + j][kk] = wv[j];
    }
    if (wid == 0) __builtin_amdgcn_s_wait_tensorcnt(0);
    __syncthreads();

    v16bf a0 = load_a_frag(&xs[wr][0],      32, 0, lane);
    v16bf a1 = load_a_frag(&xs[wr + 16][0], 32, 0, lane);
    v16bf b0 = load_b_frag(&wsT[wc][0],      32, 0, lane);
    v16bf b1 = load_b_frag(&wsT[wc + 16][0], 32, 0, lane);
    acc[0][0] = wmma_bf16(a0, b0, acc[0][0]);
    acc[0][1] = wmma_bf16(a0, b1, acc[0][1]);
    acc[1][0] = wmma_bf16(a1, b0, acc[1][0]);
    acc[1][1] = wmma_bf16(a1, b1, acc[1][1]);
    __syncthreads();
  }

  const int col = lane & 15, gh = lane >> 4;
  const float scale = 0.14433756729740643f;      // 48^-0.5
#pragma unroll
  for (int mt = 0; mt < 2; ++mt)
#pragma unroll
    for (int nt = 0; nt < 2; ++nt)
#pragma unroll
      for (int i = 0; i < 8; ++i) {
        int m = rt + wr + mt * 16 + gh * 8 + i;
        int n = ct + wc + nt * 16 + col;
        float v = acc[mt][nt][i];
        if (w == 0) {
          float t = (v + bq[n]) * scale;
          qb[(size_t)m * (NH * DP) + (n / DH) * DP + (n % DH)] = (bf16_t)t;
        } else if (w == 1) {
          kb[(size_t)m * (NH * DP) + (n / DH) * DP + (n % DH)] = (bf16_t)v;
        } else if (w == 2) {
          vb[(size_t)m * (NH * DP) + (n / DH) * DP + (n % DH)] = (bf16_t)v;
        } else {
          gate[(size_t)m * CX + n] = 1.f / (1.f + __expf(-v));
        }
      }
}

// ---------------- Kernel 2: flash attention with pair bias + gating ----------------
// grid = (SEQ/128, NH). 8 waves, each owns 16 query rows; KV streamed in 64-tiles.
// Q and K tiles staged by TDM; V staged manually (needs K-transpose for B frags).

__global__ __launch_bounds__(256) void attn_kernel(
    const bf16_t* __restrict__ qb, const bf16_t* __restrict__ kb,
    const bf16_t* __restrict__ vb, const float* __restrict__ gate,
    const float* __restrict__ pair, const unsigned char* __restrict__ mask,
    float* __restrict__ out) {
  const int qt = blockIdx.x * 128, h = blockIdx.y;

  __shared__ __align__(16) bf16_t ldsQ[128][DP];     // 16 KB (TDM target)
  __shared__ __align__(16) bf16_t ldsK[64][DP];      //  8 KB (TDM target) [kv][d]
  __shared__ __align__(16) bf16_t ldsVt[DP][64];     //  8 KB  [d][kv]
  __shared__ __align__(16) bf16_t ldsP[8][16][64];   // 16 KB  per-wave probs

  const int tid = threadIdx.x, lane = tid & 31, wid = tid >> 5;
  const int col = lane & 15, gh = lane >> 4;

  if (wid == 0) {  // DMA Q tile: 128 x 64, row pitch NH*DP (completion covered below)
    tdm_load_tile_2d(lds_off_u32(&ldsQ[0][0]),
                     qb + (size_t)qt * (NH * DP) + h * DP, (unsigned)DP, 128u,
                     (unsigned)(NH * DP));
  }

  float mrow[8], lrow[8];
  v8f acc[4];
#pragma unroll
  for (int i = 0; i < 8; ++i) { mrow[i] = -3.0e38f; lrow[i] = 0.f; }
#pragma unroll
  for (int nt = 0; nt < 4; ++nt)
#pragma unroll
    for (int i = 0; i < 8; ++i) acc[nt][i] = 0.f;

  for (int kv = 0; kv < SEQ; kv += 64) {
    __syncthreads();
    if (wid == 0) {  // DMA K tile: 64 x 64, row pitch NH*DP
      tdm_load_tile_2d(lds_off_u32(&ldsK[0][0]),
                       kb + (size_t)kv * (NH * DP) + h * DP, (unsigned)DP, 64u,
                       (unsigned)(NH * DP));
    }
    { // stage V tile transposed [d][kv]
      int r = tid >> 2, c0 = (tid & 3) * 16;
      const v8bf* src = (const v8bf*)(vb + (size_t)(kv + r) * (NH * DP) + h * DP + c0);
      v8bf v0 = src[0], v1 = src[1];
#pragma unroll
      for (int j = 0; j < 8; ++j) { ldsVt[c0 + j][r] = v0[j]; ldsVt[c0 + 8 + j][r] = v1[j]; }
    }
    if (wid == 0) __builtin_amdgcn_s_wait_tensorcnt(0);   // Q (first iter) + K complete
    __syncthreads();

    // ---- logits tile 16(q) x 64(kv): s = q . k^T  (K-dim = 64 padded head dim)
    v16bf qa0 = load_a_frag(&ldsQ[wid * 16][0], DP, 0,  lane);
    v16bf qa1 = load_a_frag(&ldsQ[wid * 16][0], DP, 32, lane);
    v8f s[4];
#pragma unroll
    for (int nt = 0; nt < 4; ++nt) {
      v16bf b0 = load_b_frag(&ldsK[nt * 16][0], DP, 0,  lane);
      v16bf b1 = load_b_frag(&ldsK[nt * 16][0], DP, 32, lane);
      v8f z;
#pragma unroll
      for (int i = 0; i < 8; ++i) z[i] = 0.f;
      z = wmma_bf16(qa0, b0, z);
      z = wmma_bf16(qa1, b1, z);
      s[nt] = z;
    }

    // ---- add pair bias (streamed NT) + mask; prefetch next KV tile of the stream
#pragma unroll
    for (int nt = 0; nt < 4; ++nt) {
      int kvg = kv + nt * 16 + col;
      bool mk = (mask[kvg] != 0);
      const float* pp = pair + ((size_t)h * SEQ + (qt + wid * 16)) * SEQ + kvg;
      if (kv + 64 < SEQ) __builtin_prefetch(pp + 64, 0, 0);
#pragma unroll
      for (int i = 0; i < 8; ++i) {
        int m = gh * 8 + i;
        float pl = __builtin_nontemporal_load(pp + (size_t)m * SEQ);
        float val = s[nt][i] + pl;
        s[nt][i] = mk ? val : NEG_BIG;
      }
    }

    // ---- online softmax (row = lane-group of 16; butterfly reductions)
#pragma unroll
    for (int i = 0; i < 8; ++i) {
      float rm = fmaxf(fmaxf(s[0][i], s[1][i]), fmaxf(s[2][i], s[3][i]));
      rm = fmaxf(rm, __shfl_xor(rm, 1));
      rm = fmaxf(rm, __shfl_xor(rm, 2));
      rm = fmaxf(rm, __shfl_xor(rm, 4));
      rm = fmaxf(rm, __shfl_xor(rm, 8));
      float mnew = fmaxf(mrow[i], rm);
      float alpha = __expf(mrow[i] - mnew);
      mrow[i] = mnew;
      float rs = 0.f;
#pragma unroll
      for (int nt = 0; nt < 4; ++nt) {
        float p = __expf(s[nt][i] - mnew);
        s[nt][i] = p;
        rs += p;
      }
      rs += __shfl_xor(rs, 1);
      rs += __shfl_xor(rs, 2);
      rs += __shfl_xor(rs, 4);
      rs += __shfl_xor(rs, 8);
      lrow[i] = lrow[i] * alpha + rs;
#pragma unroll
      for (int nt = 0; nt < 4; ++nt) acc[nt][i] *= alpha;
    }

    // ---- probs -> bf16 via per-wave LDS (re-layout C-frag -> A-frag)
#pragma unroll
    for (int nt = 0; nt < 4; ++nt)
#pragma unroll
      for (int i = 0; i < 8; ++i)
        ldsP[wid][gh * 8 + i][nt * 16 + col] = (bf16_t)s[nt][i];
    // same-wave LDS ops are in-order: no barrier needed before re-reading ldsP[wid]

    // ---- acc += P(16x64) @ V(64x64)
    v16bf pa0 = load_a_frag(&ldsP[wid][0][0], 64, 0,  lane);
    v16bf pa1 = load_a_frag(&ldsP[wid][0][0], 64, 32, lane);
#pragma unroll
    for (int nt = 0; nt < 4; ++nt) {
      v16bf b0 = load_b_frag(&ldsVt[nt * 16][0], 64, 0,  lane);
      v16bf b1 = load_b_frag(&ldsVt[nt * 16][0], 64, 32, lane);
      acc[nt] = wmma_bf16(pa0, b0, acc[nt]);
      acc[nt] = wmma_bf16(pa1, b1, acc[nt]);
    }
  }

  // ---- epilogue: out = gate * (acc / l), only d < 48 is real
#pragma unroll
  for (int nt = 0; nt < 4; ++nt) {
    int d = nt * 16 + col;
    if (d < DH) {
#pragma unroll
      for (int i = 0; i < 8; ++i) {
        int m = qt + wid * 16 + gh * 8 + i;
        size_t oi = (size_t)m * CX + h * DH + d;
        float o = acc[nt][i] / lrow[i];
        out[oi] = gate[oi] * o;
      }
    }
  }
}

// ---------------- launch ----------------

extern "C" void kernel_launch(void* const* d_in, const int* in_sizes, int n_in,
                              void* d_out, int out_size, void* d_ws, size_t ws_size,
                              hipStream_t stream) {
  const float* x            = (const float*)d_in[0];
  const unsigned char* mask = (const unsigned char*)d_in[1];
  const float* pair         = (const float*)d_in[2];
  const float* Wq           = (const float*)d_in[3];
  const float* bq           = (const float*)d_in[4];
  const float* Wk           = (const float*)d_in[5];
  const float* Wv           = (const float*)d_in[6];
  const float* Wg           = (const float*)d_in[7];
  float* out = (float*)d_out;

  char* ws = (char*)d_ws;
  bf16_t* xb   = (bf16_t*)(ws);                    // 2048*768  bf16 = 3.00 MB
  bf16_t* wb   = (bf16_t*)(ws + 3145728);          // 4*768*768 bf16 = 4.50 MB
  bf16_t* qb   = (bf16_t*)(ws + 7864320);          // 2048*1024 bf16 = 4.00 MB
  bf16_t* kb   = (bf16_t*)(ws + 12058624);         // 4.00 MB
  bf16_t* vb   = (bf16_t*)(ws + 16252928);         // 4.00 MB
  float*  gate = (float*)(ws + 20447232);          // 2048*768  f32  = 6.00 MB

  // 4*768*768 = 2359296 elements -> 9216 blocks of 256
  prep_kernel<<<9216, 256, 0, stream>>>(x, Wq, Wk, Wv, Wg, xb, wb, qb, kb, vb);
  proj_kernel<<<dim3(SEQ / 128, CX / 64, 4), 256, 0, stream>>>(xb, wb, bq, qb, kb, vb, gate);
  attn_kernel<<<dim3(SEQ / 128, NH), 256, 0, stream>>>(qb, kb, vb, gate, pair, mask, out);
}
